// HS_MSA_23218593202390
// MI455X (gfx1250) — matrix operand order repaired and checked by reading, not compiled, not run on hardware
//
#include <hip/hip_runtime.h>
#include <hip/hip_bf16.h>

typedef _Float16 half_t;
typedef __attribute__((ext_vector_type(16))) _Float16 v16h;
typedef __attribute__((ext_vector_type(8)))  float    v8f;
typedef __attribute__((ext_vector_type(4)))  unsigned int u32x4;
typedef __attribute__((ext_vector_type(8)))  int i32x8;
typedef __attribute__((ext_vector_type(4)))  int i32x4;

#define WMMA_F32_16x16x32_F16(a, b, c) \
  __builtin_amdgcn_wmma_f32_16x16x32_f16(false, (a), false, (b), (short)0, (c), false, false)

#if defined(__has_builtin)
#if __has_builtin(__builtin_amdgcn_tensor_load_to_lds) && \
    __has_builtin(__builtin_amdgcn_s_wait_tensorcnt)
#define HAVE_TDM 1
#endif
#endif
#ifndef HAVE_TDM
#define HAVE_TDM 0
#endif

#if HAVE_TDM
// Tensor Data Mover: DMA a 2D tile (rows x rowbytes, pitchbytes apart) from
// global memory into LDS. D# built per CDNA5 ISA.
// data_size = 8 bytes, so rowbytes/pitchbytes must be multiples of 8.
__device__ __forceinline__ void tdm_load_2d(const void* lds_dst, const void* gsrc,
                                            unsigned rowbytes, unsigned rows,
                                            unsigned long long pitchbytes) {
  unsigned long long ga = (unsigned long long)(size_t)gsrc;
  unsigned lds_off = (unsigned)(size_t)lds_dst;          // LDS aperture: low 32b = offset
  unsigned nx = rowbytes >> 3;                           // 8-byte elements per row
  unsigned long long st = pitchbytes >> 3;               // stride in 8B units
  u32x4 g0 = {};
  g0[0] = 1u;                                            // count=1 (valid user D#)
  g0[1] = lds_off;                                       // lds_addr
  g0[2] = (unsigned)ga;                                  // global_addr[31:0]
  g0[3] = (unsigned)((ga >> 32) & 0x01ffffffu) | 0x80000000u; // addr[56:32] | type=2
  i32x8 g1 = {};
  g1[0] = 0x00030000;                                    // wg_mask=0, data_size=3 (8B)
  g1[1] = (int)((nx & 0xffffu) << 16);                   // tensor_dim0[15:0]
  g1[2] = (int)(((nx >> 16) & 0xffffu) |                 // tensor_dim0[31:16]
                ((rows & 0xffffu) << 16));               // tensor_dim1[15:0]
  g1[3] = (int)((nx & 0xffffu) << 16);                   // tensor_dim1 hi=0 | tile_dim0
  g1[4] = (int)(rows & 0xffffu);                         // tile_dim1 (tile_dim2=0)
  g1[5] = (int)st;                                       // tensor_dim0_stride[31:0]
  g1[6] = (int)((st >> 32) & 0xffffu);                   // stride[47:32]
  g1[7] = 0;
  i32x4 g2 = {};
  i32x4 g3 = {};
#if __clang_major__ >= 23
  i32x8 g4 = {};
  __builtin_amdgcn_tensor_load_to_lds(g0, g1, g2, g3, g4, 0);
#else
  __builtin_amdgcn_tensor_load_to_lds(g0, g1, g2, g3, 0);
#endif
}
#endif

// ---- WMMA fragment helpers (contiguous k = 16*(lane/16) + j, wave32) -------------------
__device__ __forceinline__ v16h load_A_frag(const half_t* A, int lda, int lane) {
  const int row = lane & 15;
  const int kbl = (lane >> 4) << 4;   // 0 or 16
  v16h a;
#pragma unroll
  for (int j = 0; j < 16; ++j) {
    a[j] = A[row * lda + kbl + j];
  }
  return a;
}

__device__ __forceinline__ v16h load_B_frag(const half_t* B, int ldb, int lane) {
  const int col = lane & 15;
  const int kbl = (lane >> 4) << 4;   // 0 or 16
  v16h b;
#pragma unroll
  for (int j = 0; j < 16; ++j) b[j] = B[(kbl + j) * ldb + col];
  return b;
}

__device__ __forceinline__ v16h load_Bt_frag(const half_t* Bt, int ldb, int lane) {
  const int col = lane & 15;
  const int kbl = (lane >> 4) << 4;
  v16h b;
#pragma unroll
  for (int j = 0; j < 16; ++j) b[j] = Bt[col * ldb + kbl + j];
  return b;
}

// ---- Kernel 0: convert f32 weights to f16 in B-fragment order -------------
__global__ __launch_bounds__(256) void prep_weights(
    const float* __restrict__ Wq, const float* __restrict__ Wkv,
    const float* __restrict__ Wo, half_t* __restrict__ wqkv_frag,
    half_t* __restrict__ wo_frag) {
  int i = blockIdx.x * 256 + threadIdx.x;
  const int QKV = 48 * 8 * 512;
  const int WOF = 16 * 8 * 512;
  if (i < QKV) {
    int nt = i / (8 * 512);
    int r  = i % (8 * 512);
    int ks = r / 512, q = r % 512;
    int l = q >> 4, j = q & 15;
    int k = ks * 32 + ((l >> 4) << 4) + j;
    int n = nt * 16 + (l & 15);
    float v = (n < 256) ? Wq[k * 256 + n] : Wkv[k * 512 + (n - 256)];
    wqkv_frag[i] = (half_t)v;
  } else if (i < QKV + WOF) {
    int d = i - QKV;
    int nt = d / (8 * 512);
    int r  = d % (8 * 512);
    int ks = r / 512, q = r % 512;
    int l = q >> 4, j = q & 15;
    int k = ks * 32 + ((l >> 4) << 4) + j;
    int n = nt * 16 + (l & 15);
    wo_frag[d] = (half_t)Wo[k * 256 + n];
  }
}

// ---- Kernel 1: fused QKV projection, scatter into windowed token order ----
__global__ __launch_bounds__(256) void qkv_proj(
    const float* __restrict__ x, const half_t* __restrict__ wqkv_frag,
    half_t* __restrict__ qb, half_t* __restrict__ kbuf, half_t* __restrict__ vbuf) {
  extern __shared__ char smem[];
  half_t* Asm = (half_t*)smem;                       // 16 x 256 f16 = 8 KB
  const int tid  = threadIdx.x;
  const int tr16 = blockIdx.x;
  const float* xrow = x + (size_t)tr16 * 16 * 256;
  __builtin_prefetch(xrow + 16 * 256, 0, 1);
#if HAVE_TDM
  float* Xf32 = (float*)(smem + 8192);               // 16 x 256 f32 = 16 KB
  if (tid < 32) {                                    // wave 0 drives the DMA
    tdm_load_2d(Xf32, xrow, 16384, 1, 16384);
    __builtin_amdgcn_s_wait_tensorcnt(0);
  }
  __syncthreads();
#pragma unroll
  for (int it = 0; it < 16; ++it) {
    int idx = it * 256 + tid;
    Asm[idx] = (half_t)Xf32[idx];
  }
#else
#pragma unroll
  for (int it = 0; it < 16; ++it) {
    int idx = it * 256 + tid;
    Asm[idx] = (half_t)xrow[idx];
  }
#endif
  __syncthreads();
  const int wave = tid >> 5, lane = tid & 31;
  const int mb = (lane >> 4) << 3;
  const int cl = lane & 15;
#pragma unroll
  for (int t6 = 0; t6 < 6; ++t6) {
    const int nt = wave * 6 + t6;
    v8f acc = {};
#pragma unroll
    for (int ks = 0; ks < 8; ++ks) {
      v16h a = load_A_frag(Asm + ks * 32, 256, lane);
      v16h b = *(const v16h*)(wqkv_frag + (size_t)(nt * 8 + ks) * 512 + lane * 16);
      acc = WMMA_F32_16x16x32_F16(a, b, acc);
    }
    const int coln = nt * 16 + cl;
#pragma unroll
    for (int r = 0; r < 8; ++r) {
      int grow = tr16 * 16 + mb + r;
      int bb = grow >> 14, rem = grow & 16383;
      int hh = rem >> 7, ww = rem & 127;
      int t = (((bb << 8) + ((hh >> 3) << 4) + (ww >> 3)) << 6) +
              ((hh & 7) << 3) + (ww & 7);
      half_t val = (half_t)acc[r];
      if (coln < 256)      qb  [(size_t)t * 256 + coln]       = val;
      else if (coln < 512) kbuf[(size_t)t * 256 + coln - 256] = val;
      else                 vbuf[(size_t)t * 256 + coln - 512] = val;
    }
  }
}

// ---- Kernel 2: branch-1 window attention (seq=64, d=32) -------------------
__global__ __launch_bounds__(64) void attn1(
    const half_t* __restrict__ qb, const half_t* __restrict__ kbuf,
    const half_t* __restrict__ vbuf, const float* __restrict__ pos1,
    half_t* __restrict__ o12) {
  extern __shared__ char smem[];
  float*  S  = (float*)smem;                         // 64x64 f32 = 16384
  half_t* P  = (half_t*)(smem + 16384);              // 64x64 f16 =  8192
  half_t* Qs = (half_t*)(smem + 24576);              // 64x32 f16 =  4096
  half_t* Ks = (half_t*)(smem + 28672);
  half_t* Vs = (half_t*)(smem + 32768);              // total 36864 B
  const int tid = threadIdx.x;
  const int h  = blockIdx.x & 3;
  const int wn = blockIdx.x >> 2;
  const size_t t0 = (size_t)wn * 64;
  const int c0 = h * 32;
#if HAVE_TDM
  if (tid < 32) {
    const size_t base = t0 * 256 + c0;
    tdm_load_2d(Qs, qb   + base, 64, 64, 512);       // 64 rows x 64 B, pitch 512 B
    tdm_load_2d(Ks, kbuf + base, 64, 64, 512);
    tdm_load_2d(Vs, vbuf + base, 64, 64, 512);
    __builtin_amdgcn_s_wait_tensorcnt(0);
  }
#else
#pragma unroll
  for (int it = 0; it < 32; ++it) {
    int idx = it * 64 + tid;
    int row = idx >> 5, d = idx & 31;
    size_t src = (t0 + row) * 256 + c0 + d;
    Qs[idx] = qb[src];  Ks[idx] = kbuf[src];  Vs[idx] = vbuf[src];
  }
#endif
  __syncthreads();
  const int wave = tid >> 5, lane = tid & 31;
#pragma unroll
  for (int rti = 0; rti < 2; ++rti) {
    int rt = wave * 2 + rti;
    v16h a = load_A_frag(Qs + rt * 16 * 32, 32, lane);
#pragma unroll
    for (int ct = 0; ct < 4; ++ct) {
      v16h b = load_Bt_frag(Ks + ct * 16 * 32, 32, lane);
      v8f c = {};
      c = WMMA_F32_16x16x32_F16(a, b, c);
      int col = ct * 16 + (lane & 15);
      int mb = (lane >> 4) << 3;
#pragma unroll
      for (int r = 0; r < 8; ++r) S[(rt * 16 + mb + r) * 64 + col] = c[r];
    }
  }
  __syncthreads();
  {
    const float scale = 0.17677669529663689f;
    const float* pr = pos1 + (size_t)(h * 64 + tid) * 64;
    float* Sr = S + tid * 64;
    float m = -1e30f;
#pragma unroll
    for (int j = 0; j < 64; ++j) { float v = Sr[j] * scale + pr[j]; Sr[j] = v; m = fmaxf(m, v); }
    float sum = 0.f;
#pragma unroll
    for (int j = 0; j < 64; ++j) { float e = __expf(Sr[j] - m); Sr[j] = e; sum += e; }
    float inv = 1.f / sum;
    half_t* Pr = P + tid * 64;
#pragma unroll
    for (int j = 0; j < 64; ++j) Pr[j] = (half_t)(Sr[j] * inv);
  }
  __syncthreads();
#pragma unroll
  for (int rti = 0; rti < 2; ++rti) {
    int rt = wave * 2 + rti;
#pragma unroll
    for (int ct = 0; ct < 2; ++ct) {
      v8f acc = {};
#pragma unroll
      for (int kk = 0; kk < 2; ++kk) {
        v16h a = load_A_frag(P + rt * 16 * 64 + kk * 32, 64, lane);
        v16h b = load_B_frag(Vs + kk * 32 * 32 + ct * 16, 32, lane);
        acc = WMMA_F32_16x16x32_F16(a, b, acc);
      }
      int col = c0 + ct * 16 + (lane & 15);
      int mb = (lane >> 4) << 3;
#pragma unroll
      for (int r = 0; r < 8; ++r)
        o12[(t0 + rt * 16 + mb + r) * 256 + col] = (half_t)acc[r];
    }
  }
}

// ---- Kernel 3: branch-2 grid attention (seq=256, d=32), streaming softmax -
__global__ __launch_bounds__(256) void attn2(
    const half_t* __restrict__ qb, const half_t* __restrict__ kbuf,
    const half_t* __restrict__ vbuf, const float* __restrict__ pos2,
    half_t* __restrict__ o12) {
  extern __shared__ char smem[];
  float*  S    = (float*)smem;                       // 256x64 f32 = 65536
  float*  Obuf = (float*)(smem + 65536);             // 256x32 f32 = 32768
  half_t* P    = (half_t*)(smem + 98304);            // 256x64 f16 = 32768
  half_t* Qs   = (half_t*)(smem + 131072);           // 256x32 f16 = 16384
  half_t* Kc   = (half_t*)(smem + 147456);           //  64x32 f16 =  4096
  half_t* Vc   = (half_t*)(smem + 151552);           // total 155648 B
  const int tid = threadIdx.x;
  const int h  = blockIdx.x & 3;
  const int mm = (blockIdx.x >> 2) & 63;
  const int bI = blockIdx.x >> 8;
  const int c0 = 128 + h * 32;
#if HAVE_TDM
  if (tid < 32) {
    size_t tq = (((size_t)(bI * 256)) << 6) + mm;    // token of seq pos 0
    tdm_load_2d(Qs, qb + tq * 256 + c0, 64, 256, 32768); // 256 rows x 64 B
    __builtin_amdgcn_s_wait_tensorcnt(0);
  }
#else
#pragma unroll
  for (int it = 0; it < 32; ++it) {
    int idx = it * 256 + tid;
    int n = idx >> 5, d = idx & 31;
    size_t t = ((size_t)(bI * 256 + n) << 6) + mm;
    Qs[idx] = qb[t * 256 + c0 + d];
  }
#endif
  {
    float* Or = Obuf + tid * 32;
#pragma unroll
    for (int d = 0; d < 32; ++d) Or[d] = 0.f;
  }
  float m_run = -1e30f, l_run = 0.f;
  const int wave = tid >> 5, lane = tid & 31;
  const float scale = 0.17677669529663689f;
  for (int chunk = 0; chunk < 4; ++chunk) {
    __syncthreads();
#if HAVE_TDM
    if (tid < 32) {
      size_t tc = (((size_t)(bI * 256 + chunk * 64)) << 6) + mm;
      tdm_load_2d(Kc, kbuf + tc * 256 + c0, 64, 64, 32768);
      tdm_load_2d(Vc, vbuf + tc * 256 + c0, 64, 64, 32768);
      __builtin_amdgcn_s_wait_tensorcnt(0);
    }
#else
#pragma unroll
    for (int it = 0; it < 8; ++it) {
      int idx = it * 256 + tid;
      int n = idx >> 5, d = idx & 31;
      size_t t = ((size_t)(bI * 256 + chunk * 64 + n) << 6) + mm;
      Kc[idx] = kbuf[t * 256 + c0 + d];
      Vc[idx] = vbuf[t * 256 + c0 + d];
    }
#endif
    __syncthreads();
#pragma unroll
    for (int rti = 0; rti < 2; ++rti) {
      int rt = wave * 2 + rti;
      v16h a = load_A_frag(Qs + rt * 16 * 32, 32, lane);
#pragma unroll
      for (int ct = 0; ct < 4; ++ct) {
        v16h b = load_Bt_frag(Kc + ct * 16 * 32, 32, lane);
        v8f c = {};
        c = WMMA_F32_16x16x32_F16(a, b, c);
        int col = ct * 16 + (lane & 15);
        int mb = (lane >> 4) << 3;
#pragma unroll
        for (int r = 0; r < 8; ++r) S[(rt * 16 + mb + r) * 64 + col] = c[r];
      }
    }
    __syncthreads();
    {
      const float* pr = pos2 + (size_t)(h * 256 + tid) * 256 + chunk * 64;
      float* Sr = S + tid * 64;
      float mloc = -1e30f;
#pragma unroll
      for (int j = 0; j < 64; ++j) { float v = Sr[j] * scale + pr[j]; Sr[j] = v; mloc = fmaxf(mloc, v); }
      float mnew = fmaxf(m_run, mloc);
      float corr = __expf(m_run - mnew);
      float s = 0.f;
      half_t* Pr = P + tid * 64;
#pragma unroll
      for (int j = 0; j < 64; ++j) { float e = __expf(Sr[j] - mnew); Pr[j] = (half_t)e; s += e; }
      l_run = l_run * corr + s;
      float* Or = Obuf + tid * 32;
#pragma unroll
      for (int d = 0; d < 32; ++d) Or[d] *= corr;
      m_run = mnew;
    }
    __syncthreads();
#pragma unroll
    for (int rti = 0; rti < 2; ++rti) {
      int rt = wave * 2 + rti;
#pragma unroll
      for (int ct = 0; ct < 2; ++ct) {
        v8f acc = {};
#pragma unroll
        for (int kk = 0; kk < 2; ++kk) {
          v16h a = load_A_frag(P + rt * 16 * 64 + kk * 32, 64, lane);
          v16h b = load_B_frag(Vc + kk * 32 * 32 + ct * 16, 32, lane);
          acc = WMMA_F32_16x16x32_F16(a, b, acc);
        }
        int col = ct * 16 + (lane & 15);
        int mb = (lane >> 4) << 3;
#pragma unroll
        for (int r = 0; r < 8; ++r)
          Obuf[(rt * 16 + mb + r) * 32 + col] += acc[r];
      }
    }
  }
  __syncthreads();
  {
    float inv = 1.f / l_run;
    size_t t = ((size_t)(bI * 256 + tid) << 6) + mm;
    float* Or = Obuf + tid * 32;
#pragma unroll
    for (int d = 0; d < 32; ++d)
      o12[t * 256 + c0 + d] = (half_t)(Or[d] * inv);
  }
}

// ---- Kernel 4: output projection + bias + un-window -----------------------
__global__ __launch_bounds__(256) void out_proj(
    const half_t* __restrict__ o12, const half_t* __restrict__ wo_frag,
    const float* __restrict__ bo, float* __restrict__ out) {
  extern __shared__ char smem[];
  half_t* Asm = (half_t*)smem;                       // 16 x 256 f16 = 8 KB
  const int tid  = threadIdx.x;
  const int tr16 = blockIdx.x;
  const half_t* src = o12 + (size_t)tr16 * 16 * 256;
#if HAVE_TDM
  if (tid < 32) {
    tdm_load_2d(Asm, src, 8192, 1, 8192);            // flat 8 KB tile copy
    __builtin_amdgcn_s_wait_tensorcnt(0);
  }
#else
#pragma unroll
  for (int it = 0; it < 16; ++it) {
    int idx = it * 256 + tid;
    Asm[idx] = src[idx];
  }
#endif
  __syncthreads();
  const int wave = tid >> 5, lane = tid & 31;
  const int mb = (lane >> 4) << 3;
#pragma unroll
  for (int sub = 0; sub < 2; ++sub) {
    const int nt = wave * 2 + sub;
    v8f acc = {};
#pragma unroll
    for (int ks = 0; ks < 8; ++ks) {
      v16h a = load_A_frag(Asm + ks * 32, 256, lane);
      v16h b = *(const v16h*)(wo_frag + (size_t)(nt * 8 + ks) * 512 + lane * 16);
      acc = WMMA_F32_16x16x32_F16(a, b, acc);
    }
    const int coln = nt * 16 + (lane & 15);
    const float bias = bo[coln];
#pragma unroll
    for (int r = 0; r < 8; ++r) {
      int t  = tr16 * 16 + mb + r;
      int bb = t >> 14, rr = t & 16383;
      int n = rr >> 6, mmw = rr & 63;
      int hh = ((n >> 4) << 3) + (mmw >> 3);
      int ww = ((n & 15) << 3) + (mmw & 7);
      out[((size_t)((bb * 128 + hh) * 128 + ww)) * 256 + coln] = acc[r] + bias;
    }
  }
}

extern "C" void kernel_launch(void* const* d_in, const int* in_sizes, int n_in,
                              void* d_out, int out_size, void* d_ws, size_t ws_size,
                              hipStream_t stream) {
  (void)in_sizes; (void)n_in; (void)out_size; (void)ws_size;
  const float* x    = (const float*)d_in[0];
  const float* Wq   = (const float*)d_in[1];
  const float* Wkv  = (const float*)d_in[2];
  const float* Wo   = (const float*)d_in[3];
  const float* bo   = (const float*)d_in[4];
  const float* pos1 = (const float*)d_in[5];
  const float* pos2 = (const float*)d_in[6];
  float* out = (float*)d_out;

  char* ws = (char*)d_ws;
  const size_t QKVF = 393216;
  const size_t WOF  = 131072;
  const size_t TOK  = (size_t)65536 * 256 * sizeof(half_t);   // 32 MB
  half_t* wqkv_frag = (half_t*)(ws);
  half_t* wo_frag   = (half_t*)(ws + QKVF);
  half_t* qb   = (half_t*)(ws + QKVF + WOF);
  half_t* kbuf = (half_t*)(ws + QKVF + WOF + TOK);
  half_t* vbuf = (half_t*)(ws + QKVF + WOF + 2 * TOK);
  half_t* o12  = (half_t*)(ws + QKVF + WOF + 3 * TOK);

  prep_weights<<<1024, 256, 0, stream>>>(Wq, Wkv, Wo, wqkv_frag, wo_frag);
  qkv_proj<<<4096, 256, 24576, stream>>>(x, wqkv_frag, qb, kbuf, vbuf);
  attn1<<<4096, 64, 36864, stream>>>(qb, kbuf, vbuf, pos1, o12);
  attn2<<<1024, 256, 155648, stream>>>(qb, kbuf, vbuf, pos2, o12);
  out_proj<<<4096, 256, 8192, stream>>>(o12, wo_frag, bo, out);
}
